// Encoder_2886218023678
// MI455X (gfx1250) — compile-verified, matrix operand order checked by reading, not run on hardware
//
#include <hip/hip_runtime.h>
#include <hip/hip_bf16.h>
#include <math.h>

// ---------------- problem constants (from reference) ----------------
#define B_       16
#define T_TOTAL_ 16
#define HIST_    8
#define C_       1000
#define IN_DIM_  16
#define EMB_     32
#define HID_     64
#define E_       16000
#define N_       (B_ * C_)     // 16000
#define EB_      (E_ * B_)     // 256000
#define IN_CONV_ 49
#define KPA_     64            // IN_CONV padded for WMMA K
#define IN_GRU_  113
#define KPG_     128           // IN_GRU padded for WMMA K
#define G3_      192           // 3*HID

#define CEIL_DIV(a,b) (((a)+(b)-1)/(b))

typedef __attribute__((ext_vector_type(16))) _Float16 v16h;
typedef __attribute__((ext_vector_type(8)))  _Float16 v8h;
typedef __attribute__((ext_vector_type(8)))  float    v8f;

// ---------------- WMMA helpers (CDNA5 wave32, 16x16x32 f16 -> f32) ----------------
// A operand (16x32 f16): lane L holds row (L&15); K chunks [hi*8,hi*8+8) and
// [16+hi*8, 16+hi*8+8) with hi = L>>4.  With row-major f16 memory these are two
// contiguous 16B loads.  B operand (32x16) mirrors this with column = L&15 when
// the tile is stored column-major-by-N ([n][k]).  So one loader serves both.
__device__ __forceinline__ v16h load_op16(const _Float16* p) {
  v8h lo = *reinterpret_cast<const v8h*>(p);
  v8h hi = *reinterpret_cast<const v8h*>(p + 16);
  return __builtin_shufflevector(lo, hi, 0,1,2,3,4,5,6,7,8,9,10,11,12,13,14,15);
}

__device__ __forceinline__ v8f wmma16(v16h a, v16h b, v8f c) {
  // (neg_a, A, neg_b, B, c_mod, C, reuse_a, reuse_b)
  return __builtin_amdgcn_wmma_f32_16x16x32_f16(false, a, false, b, (short)0, c,
                                                false, false);
}

__device__ __forceinline__ float sigmoidf_(float x) { return 1.0f / (1.0f + expf(-x)); }

// monotone float<->uint encoding for atomic segment-max
__device__ __forceinline__ unsigned enc_f(float f) {
  unsigned u = __float_as_uint(f);
  return (u & 0x80000000u) ? ~u : (u | 0x80000000u);
}
__device__ __forceinline__ float dec_f(unsigned u) {
  return (u & 0x80000000u) ? __uint_as_float(u ^ 0x80000000u) : __uint_as_float(~u);
}

// ---------------- weight packing ----------------
// Pack W (f32) into f16 B-operand tiles: tile (nt,kt) has 512 f16 laid out
// [n(16)][k(32)].  transposed==0: W is [fin][fout] (use W[k][n]);
// transposed==1: W is [fout][fin] (use W[n][k], i.e. effective W^T).
__global__ void pack_b_kernel(const float* __restrict__ W, _Float16* __restrict__ PB,
                              int fin, int fout, int KP, int transposed) {
  int i = blockIdx.x * blockDim.x + threadIdx.x;
  int total = KP * fout;
  if (i >= total) return;
  int k = i % KP;
  int n = i / KP;
  int nt = n >> 4, nn = n & 15, kt = k >> 5, kk2 = k & 31;
  int KT2 = KP >> 5;
  float v = 0.0f;
  if (k < fin) v = transposed ? W[(size_t)n * fin + k] : W[(size_t)k * fout + n];
  PB[((size_t)(nt * KT2 + kt)) * 512 + nn * 32 + kk2] = (_Float16)v;
}

__global__ void zero_f32_kernel(float* p, int n) {
  int i = blockIdx.x * blockDim.x + threadIdx.x;
  if (i < n) p[i] = 0.0f;
}

__global__ void init_seg_kernel(unsigned* m_enc, float* s) {
  int i = blockIdx.x * blockDim.x + threadIdx.x;
  if (i < N_) { m_enc[i] = 0u; s[i] = 0.0f; }
}

// ---------------- step kernel 1: node features + wind ----------------
__global__ void build_features_kernel(const float* __restrict__ X, const float* __restrict__ y,
                                      const float* __restrict__ emb_table,
                                      const float* __restrict__ xn_prev,
                                      _Float16* __restrict__ xf_h, float* __restrict__ wind,
                                      int t) {
  int idx = blockIdx.x * blockDim.x + threadIdx.x;
  if (idx >= N_) return;
  int b = idx / C_, c = idx - b * C_;
  const float* xr = X + (((size_t)b * T_TOTAL_ + t) * C_ + c) * IN_DIM_;
  int id = (int)xr[IN_DIM_ - 1];
  _Float16* o = xf_h + (size_t)idx * KPA_;
  o[0] = (_Float16)xn_prev[idx];
  o[1] = (_Float16)y[((size_t)b * T_TOTAL_ + t) * C_ + c];
#pragma unroll
  for (int j = 0; j < IN_DIM_ - 1; ++j) o[2 + j] = (_Float16)xr[j];
  const float* er = emb_table + (size_t)id * EMB_;
#pragma unroll
  for (int k = 0; k < EMB_; ++k) o[2 + (IN_DIM_ - 1) + k] = (_Float16)er[k];
#pragma unroll
  for (int k = IN_CONV_; k < KPA_; ++k) o[k] = (_Float16)0.0f;
  // wind features (u10 = feats[-2]*3+0.5, v10 = feats[-1]*3-0.3)
  float u10 = xr[13] * 3.0f + 0.5f;
  float v10 = xr[14] * 3.0f - 0.3f;
  float spd = hypotf(u10, v10);
  float d = 1.5707963267948966f - atan2f(-v10, -u10);
  if (d <= 0.0f) d += 6.283185307179586f;
  if (spd == 0.0f) d = 0.0f;
  wind[idx * 2 + 0] = spd;
  wind[idx * 2 + 1] = d;
}

// ---------------- step kernel 2: q/k/v/skip GEMMs via WMMA ----------------
// grid = 1000 (mtile), block = 128 (4 waves, wave = ntile of HID=64)
__global__ __launch_bounds__(128)
void gemm_qkvs_kernel(const _Float16* __restrict__ xf_h,
                      const _Float16* __restrict__ PBq, const _Float16* __restrict__ PBk,
                      const _Float16* __restrict__ PBv, const _Float16* __restrict__ PBs,
                      const float* __restrict__ bq, const float* __restrict__ bk,
                      const float* __restrict__ bv, const float* __restrict__ bs,
                      float* __restrict__ q, float* __restrict__ kk,
                      float* __restrict__ vv, float* __restrict__ outb) {
  int mtile = blockIdx.x;
  int wave = threadIdx.x >> 5;        // ntile 0..3
  int lane = threadIdx.x & 31;
  int idx16 = lane & 15, hi = lane >> 4;
  int row = mtile * 16 + idx16;
  const _Float16* arow = xf_h + (size_t)row * KPA_;
  v16h a0 = load_op16(arow + 0 * 32 + hi * 8);   // kt=0
  v16h a1 = load_op16(arow + 1 * 32 + hi * 8);   // kt=1

  const _Float16* PBs4[4] = {PBq, PBk, PBv, PBs};
  const float*    bias4[4] = {bq, bk, bv, bs};
  float*          out4[4] = {q, kk, vv, outb};
#pragma unroll
  for (int m = 0; m < 4; ++m) {
    const _Float16* tb = PBs4[m];
    v16h b0 = load_op16(tb + (size_t)(wave * 2 + 0) * 512 + idx16 * 32 + hi * 8);
    v16h b1 = load_op16(tb + (size_t)(wave * 2 + 1) * 512 + idx16 * 32 + hi * 8);
    v8f acc = {0.f,0.f,0.f,0.f,0.f,0.f,0.f,0.f};
    acc = wmma16(a0, b0, acc);
    acc = wmma16(a1, b1, acc);
    int col = wave * 16 + idx16;
    float bia = bias4[m][col];
#pragma unroll
    for (int r = 0; r < 8; ++r) {
      int orow = mtile * 16 + hi * 8 + r;
      out4[m][(size_t)orow * HID_ + col] = acc[r] + bia;
    }
  }
}

// ---------------- edge helpers ----------------
// Reference quirk preserved: SRC/DST use flat index i = b*E + e, while the
// edge-attr/e_emb rows use i = e2*B + b2.  Both decompositions of the same i.
__device__ __forceinline__ void edge_attr5(const int* __restrict__ ei,
                                           const float* __restrict__ eattr0,
                                           const float* __restrict__ wind,
                                           int i, float at[5]) {
  int e2 = i / B_, b2 = i - e2 * B_;
  int s2 = ei[e2];  // edge_indices[0][e2]
  float dist = eattr0[s2 * 2 + 0];
  float dire = eattr0[s2 * 2 + 1];
  float spd = wind[((size_t)b2 * C_ + s2) * 2 + 0];
  float wd  = wind[((size_t)b2 * C_ + s2) * 2 + 1];
  float th = fabsf(dire - wd);
  float adv = fmaxf(0.0f, 3.0f * spd * cosf(th) / dist);
  at[0] = dist; at[1] = dire; at[2] = spd; at[3] = wd; at[4] = adv;
}

// pass A: alpha + segment max
__global__ void edge_alpha_kernel(const int* __restrict__ ei, const float* __restrict__ eattr0,
                                  const float* __restrict__ We, const float* __restrict__ be,
                                  const float* __restrict__ q, const float* __restrict__ kk,
                                  const float* __restrict__ wind,
                                  float* __restrict__ alpha, unsigned* __restrict__ m_enc) {
  int i = blockIdx.x * blockDim.x + threadIdx.x;
  if (i >= EB_) return;
  int b = i / E_, e = i - b * E_;
  int SRC = b * C_ + ei[e];
  int DST = b * C_ + ei[E_ + e];
  float at[5];
  edge_attr5(ei, eattr0, wind, i, at);
  const float* qr = q + (size_t)DST * HID_;
  const float* kr = kk + (size_t)SRC * HID_;
  float acc = 0.0f;
  for (int h = 0; h < HID_; ++h) {
    float ee = be[h];
#pragma unroll
    for (int j = 0; j < 5; ++j) ee += at[j] * We[j * HID_ + h];
    acc += qr[h] * (kr[h] + ee);
  }
  float al = acc * 0.125f;  // 1/sqrt(64)
  alpha[i] = al;
  atomicMax(&m_enc[DST], enc_f(al));
}

// pass B: a = exp(alpha - m[dst]); segment sum  (a written in place of alpha)
__global__ void edge_expsum_kernel(const int* __restrict__ ei, float* __restrict__ alpha,
                                   const unsigned* __restrict__ m_enc, float* __restrict__ s) {
  int i = blockIdx.x * blockDim.x + threadIdx.x;
  if (i >= EB_) return;
  int b = i / E_, e = i - b * E_;
  int DST = b * C_ + ei[E_ + e];
  float a = expf(alpha[i] - dec_f(m_enc[DST]));
  alpha[i] = a;
  atomicAdd(&s[DST], a);
}

// pass C: out[dst] += (a/s[dst]) * (v[src] + e_emb)
__global__ void edge_agg_kernel(const int* __restrict__ ei, const float* __restrict__ eattr0,
                                const float* __restrict__ We, const float* __restrict__ be,
                                const float* __restrict__ vv, const float* __restrict__ wind,
                                const float* __restrict__ a, const float* __restrict__ s,
                                float* __restrict__ outb) {
  int i = blockIdx.x * blockDim.x + threadIdx.x;
  if (i >= EB_) return;
  int b = i / E_, e = i - b * E_;
  int SRC = b * C_ + ei[e];
  int DST = b * C_ + ei[E_ + e];
  float at[5];
  edge_attr5(ei, eattr0, wind, i, at);
  float p = a[i] / s[DST];
  const float* vr = vv + (size_t)SRC * HID_;
  float* orow = outb + (size_t)DST * HID_;
  for (int h = 0; h < HID_; ++h) {
    float ee = be[h];
#pragma unroll
    for (int j = 0; j < 5; ++j) ee += at[j] * We[j * HID_ + h];
    atomicAdd(&orow[h], p * (vr[h] + ee));
  }
}

// ---------------- step kernel: sigmoid + build GRU inputs ----------------
__global__ void post_gcn_kernel(const _Float16* __restrict__ xf_h, const float* __restrict__ outb,
                                const float* __restrict__ hn,
                                _Float16* __restrict__ x2_h, _Float16* __restrict__ h_h) {
  int idx = blockIdx.x * blockDim.x + threadIdx.x;
  if (idx >= N_) return;
  _Float16* x2 = x2_h + (size_t)idx * KPG_;
  const _Float16* xf = xf_h + (size_t)idx * KPA_;
#pragma unroll
  for (int j = 0; j < IN_CONV_; ++j) x2[j] = xf[j];
  const float* orow = outb + (size_t)idx * HID_;
#pragma unroll
  for (int h = 0; h < HID_; ++h) x2[IN_CONV_ + h] = (_Float16)sigmoidf_(orow[h]);
#pragma unroll
  for (int j = IN_GRU_; j < KPG_; ++j) x2[j] = (_Float16)0.0f;
  const float* hr = hn + (size_t)idx * HID_;
  _Float16* ho = h_h + (size_t)idx * HID_;
#pragma unroll
  for (int h = 0; h < HID_; ++h) ho[h] = (_Float16)hr[h];
}

// ---------------- fused GRU: 2 WMMA GEMMs + gating + output head ----------------
// grid = 1000 (mtile = 16 rows), block = 384 (12 waves, wave = ntile of 192)
__global__ __launch_bounds__(384)
void gru_kernel(const _Float16* __restrict__ x2_h, const _Float16* __restrict__ h_h,
                const _Float16* __restrict__ PBih, const _Float16* __restrict__ PBhh,
                const float* __restrict__ b_ih, const float* __restrict__ b_hh,
                const float* __restrict__ Wout, const float* __restrict__ bout,
                float* __restrict__ hn, float* __restrict__ xn) {
  __shared__ float s_gi[16][G3_];
  __shared__ float s_gh[16][G3_];
  __shared__ float s_h[16][HID_];
  int mtile = blockIdx.x;
  int nt = threadIdx.x >> 5;          // 0..11
  int lane = threadIdx.x & 31;
  int idx16 = lane & 15, hi = lane >> 4;
  int row = mtile * 16 + idx16;
  int col = nt * 16 + idx16;

  // gi = x2 @ W_ih^T + b_ih  (K = 128, 4 k-tiles)
  {
    const _Float16* arow = x2_h + (size_t)row * KPG_;
    v8f acc = {0.f,0.f,0.f,0.f,0.f,0.f,0.f,0.f};
#pragma unroll
    for (int kt = 0; kt < 4; ++kt) {
      v16h a = load_op16(arow + kt * 32 + hi * 8);
      v16h b = load_op16(PBih + (size_t)(nt * 4 + kt) * 512 + idx16 * 32 + hi * 8);
      acc = wmma16(a, b, acc);
    }
    float bia = b_ih[col];
#pragma unroll
    for (int r = 0; r < 8; ++r) s_gi[hi * 8 + r][col] = acc[r] + bia;
  }
  // gh = h @ W_hh^T + b_hh  (K = 64, 2 k-tiles)
  {
    const _Float16* arow = h_h + (size_t)row * HID_;
    v8f acc = {0.f,0.f,0.f,0.f,0.f,0.f,0.f,0.f};
#pragma unroll
    for (int kt = 0; kt < 2; ++kt) {
      v16h a = load_op16(arow + kt * 32 + hi * 8);
      v16h b = load_op16(PBhh + (size_t)(nt * 2 + kt) * 512 + idx16 * 32 + hi * 8);
      acc = wmma16(a, b, acc);
    }
    float bia = b_hh[col];
#pragma unroll
    for (int r = 0; r < 8; ++r) s_gh[hi * 8 + r][col] = acc[r] + bia;
  }
  __syncthreads();

  // GRU gating: r = sig(gi0+gh0); z = sig(gi1+gh1); n = tanh(gi2 + r*gh2)
  for (int e0 = threadIdx.x; e0 < 16 * HID_; e0 += 384) {
    int r0 = e0 >> 6, hcol = e0 & 63;
    int grow = mtile * 16 + r0;
    float rg = sigmoidf_(s_gi[r0][hcol] + s_gh[r0][hcol]);
    float zg = sigmoidf_(s_gi[r0][64 + hcol] + s_gh[r0][64 + hcol]);
    float ng = tanhf(s_gi[r0][128 + hcol] + rg * s_gh[r0][128 + hcol]);
    float hold = hn[(size_t)grow * HID_ + hcol];
    float hnew = (1.0f - zg) * ng + zg * hold;
    hn[(size_t)grow * HID_ + hcol] = hnew;
    s_h[r0][hcol] = hnew;
  }
  __syncthreads();

  // output head: xn = h_new @ Wout + bout (one thread per row)
  if (threadIdx.x < 16) {
    int r0 = threadIdx.x;
    float acc = bout[0];
#pragma unroll
    for (int h = 0; h < HID_; ++h) acc += s_h[r0][h] * Wout[h];
    xn[mtile * 16 + r0] = acc;
  }
}

__global__ void copy_out_kernel(const float* __restrict__ hn, const float* __restrict__ xn,
                                float* __restrict__ out) {
  int i = blockIdx.x * blockDim.x + threadIdx.x;
  if (i < N_ * HID_) out[i] = hn[i];
  else if (i < N_ * HID_ + N_) out[i] = xn[i - N_ * HID_];
}

// ---------------- host side ----------------
extern "C" void kernel_launch(void* const* d_in, const int* in_sizes, int n_in,
                              void* d_out, int out_size, void* d_ws, size_t ws_size,
                              hipStream_t stream) {
  const float* X       = (const float*)d_in[0];
  const float* y       = (const float*)d_in[1];
  const float* eattr0  = (const float*)d_in[2];
  const float* emb     = (const float*)d_in[3];
  const float* Wq      = (const float*)d_in[4];
  const float* bq      = (const float*)d_in[5];
  const float* Wk      = (const float*)d_in[6];
  const float* bk      = (const float*)d_in[7];
  const float* Wv      = (const float*)d_in[8];
  const float* bv      = (const float*)d_in[9];
  const float* We      = (const float*)d_in[10];
  const float* be      = (const float*)d_in[11];
  const float* Wskip   = (const float*)d_in[12];
  const float* bskip   = (const float*)d_in[13];
  const float* W_ih    = (const float*)d_in[14];
  const float* b_ih    = (const float*)d_in[15];
  const float* W_hh    = (const float*)d_in[16];
  const float* b_hh    = (const float*)d_in[17];
  const float* Wout    = (const float*)d_in[18];
  const float* bout    = (const float*)d_in[19];
  const int*   ei      = (const int*)d_in[20];

  // workspace carve (256B aligned)
  char* base = (char*)d_ws;
  size_t off = 0;
  auto carve = [&](size_t bytes) -> void* {
    void* p = base + off;
    off = (off + bytes + 255) & ~(size_t)255;
    return p;
  };
  _Float16* xf_h  = (_Float16*)carve((size_t)N_ * KPA_ * 2);
  float*    wind  = (float*)   carve((size_t)N_ * 2 * 4);
  float*    q     = (float*)   carve((size_t)N_ * HID_ * 4);
  float*    kk    = (float*)   carve((size_t)N_ * HID_ * 4);
  float*    vv    = (float*)   carve((size_t)N_ * HID_ * 4);
  float*    outb  = (float*)   carve((size_t)N_ * HID_ * 4);
  float*    alpha = (float*)   carve((size_t)EB_ * 4);
  unsigned* m_enc = (unsigned*)carve((size_t)N_ * 4);
  float*    sseg  = (float*)   carve((size_t)N_ * 4);
  _Float16* x2_h  = (_Float16*)carve((size_t)N_ * KPG_ * 2);
  _Float16* h_h   = (_Float16*)carve((size_t)N_ * HID_ * 2);
  float*    hn    = (float*)   carve((size_t)N_ * HID_ * 4);
  float*    xn    = (float*)   carve((size_t)N_ * 4);
  _Float16* PBq   = (_Float16*)carve((size_t)KPA_ * HID_ * 2);
  _Float16* PBk   = (_Float16*)carve((size_t)KPA_ * HID_ * 2);
  _Float16* PBv   = (_Float16*)carve((size_t)KPA_ * HID_ * 2);
  _Float16* PBsk  = (_Float16*)carve((size_t)KPA_ * HID_ * 2);
  _Float16* PBih  = (_Float16*)carve((size_t)KPG_ * G3_ * 2);
  _Float16* PBhh  = (_Float16*)carve((size_t)HID_ * G3_ * 2);

  // pack weights to WMMA B-operand tiles (f16)
  pack_b_kernel<<<CEIL_DIV(KPA_*HID_,256),256,0,stream>>>(Wq, PBq, IN_CONV_, HID_, KPA_, 0);
  pack_b_kernel<<<CEIL_DIV(KPA_*HID_,256),256,0,stream>>>(Wk, PBk, IN_CONV_, HID_, KPA_, 0);
  pack_b_kernel<<<CEIL_DIV(KPA_*HID_,256),256,0,stream>>>(Wv, PBv, IN_CONV_, HID_, KPA_, 0);
  pack_b_kernel<<<CEIL_DIV(KPA_*HID_,256),256,0,stream>>>(Wskip, PBsk, IN_CONV_, HID_, KPA_, 0);
  pack_b_kernel<<<CEIL_DIV(KPG_*G3_,256),256,0,stream>>>(W_ih, PBih, IN_GRU_, G3_, KPG_, 1);
  pack_b_kernel<<<CEIL_DIV(HID_*G3_,256),256,0,stream>>>(W_hh, PBhh, HID_, G3_, HID_, 1);

  // zero carries
  zero_f32_kernel<<<CEIL_DIV(N_*HID_,256),256,0,stream>>>(hn, N_*HID_);
  zero_f32_kernel<<<CEIL_DIV(N_,256),256,0,stream>>>(xn, N_);

  for (int t = 0; t < HIST_; ++t) {
    build_features_kernel<<<CEIL_DIV(N_,256),256,0,stream>>>(X, y, emb, xn, xf_h, wind, t);
    gemm_qkvs_kernel<<<N_/16,128,0,stream>>>(xf_h, PBq, PBk, PBv, PBsk,
                                             bq, bk, bv, bskip, q, kk, vv, outb);
    init_seg_kernel<<<CEIL_DIV(N_,256),256,0,stream>>>(m_enc, sseg);
    edge_alpha_kernel<<<CEIL_DIV(EB_,256),256,0,stream>>>(ei, eattr0, We, be, q, kk,
                                                          wind, alpha, m_enc);
    edge_expsum_kernel<<<CEIL_DIV(EB_,256),256,0,stream>>>(ei, alpha, m_enc, sseg);
    edge_agg_kernel<<<CEIL_DIV(EB_,256),256,0,stream>>>(ei, eattr0, We, be, vv, wind,
                                                        alpha, sseg, outb);
    post_gcn_kernel<<<CEIL_DIV(N_,256),256,0,stream>>>(xf_h, outb, hn, x2_h, h_h);
    gru_kernel<<<N_/16,384,0,stream>>>(x2_h, h_h, PBih, PBhh, b_ih, b_hh,
                                       Wout, bout, hn, xn);
  }
  copy_out_kernel<<<CEIL_DIV(N_*HID_ + N_,256),256,0,stream>>>(hn, xn, (float*)d_out);
  (void)in_sizes; (void)n_in; (void)out_size; (void)ws_size;
}
